// VoxelBackBone8x_79620103733507
// MI455X (gfx1250) — compile-verified, matrix operand order checked by reading, not run on hardware
//
#include <hip/hip_runtime.h>
#include <stdint.h>

// ---------------------------------------------------------------------------
// CDNA5 (gfx1250) sparse 3D conv backbone via wave32 WMMA bf16 gather-GEMM.
//   A fragment: 16 active voxels x 32 K (cin, zero-padded)    (bf16)
//   B fragment: 32 K x 16 cout, pre-packed to lane layout     (bf16)
//   C/D:        16 x 16 f32 accumulators; NN cout-tiles kept in registers per
//               wave so each gathered A fragment feeds NN WMMAs (amortizes the
//               bandwidth-limiting gather traffic 2-8x on the heavy layers).
// ---------------------------------------------------------------------------

typedef __attribute__((ext_vector_type(16))) __bf16 v16bf;
typedef __attribute__((ext_vector_type(8)))  float  v8f;

static __device__ __forceinline__ unsigned short f2bf(float f) {
  union { float f; unsigned u; } v; v.f = f;
  return (unsigned short)((v.u + 0x7FFFu + ((v.u >> 16) & 1u)) >> 16);
}

// ----------------------------- utility kernels -----------------------------

__global__ void k_zero(uint4* __restrict__ p, long long n16) {
  long long i = (long long)blockIdx.x * blockDim.x + threadIdx.x;
  if (i < n16) p[i] = make_uint4(0u, 0u, 0u, 0u);
}

// dedup scatter pass 1: winner = max voxel index per cell (deterministic)
__global__ void k_scatter1(const int* __restrict__ coords, int* __restrict__ slot,
                           int n, int H, int W) {
  int i = blockIdx.x * blockDim.x + threadIdx.x;
  if (i >= n) return;
  int z = coords[i * 4 + 1], y = coords[i * 4 + 2], x = coords[i * 4 + 3];
  long long lin = ((long long)z * H + y) * W + x;
  atomicMax(&slot[lin], i + 1);
}

// dedup scatter pass 2: winner writes bf16 features + mask byte
__global__ void k_scatter2(const float* __restrict__ feats, const int* __restrict__ coords,
                           const int* __restrict__ slot, unsigned short* __restrict__ t,
                           unsigned char* __restrict__ mask, int n, int H, int W, int cpad) {
  int i = blockIdx.x * blockDim.x + threadIdx.x;
  if (i >= n) return;
  int z = coords[i * 4 + 1], y = coords[i * 4 + 2], x = coords[i * 4 + 3];
  long long lin = ((long long)z * H + y) * W + x;
  if (slot[lin] != i + 1) return;
  mask[lin] = 1;
  unsigned short* d = t + lin * cpad;
  for (int c = 0; c < 4; ++c) d[c] = f2bf(feats[i * 4 + c]);
}

__global__ void k_build_list(const unsigned char* __restrict__ mask, int nvox,
                             int* __restrict__ list, int* __restrict__ count) {
  int i = blockIdx.x * blockDim.x + threadIdx.x;
  if (i >= nvox) return;
  if (mask[i]) { int p = atomicAdd(count, 1); list[p] = i; }
}

// spconv mask dilation: out active iff any in-window input active
__global__ void k_dilate(const unsigned char* __restrict__ mi, int Di, int Hi, int Wi,
                         unsigned char* __restrict__ mo, int Do, int Ho, int Wo,
                         int sz, int sy, int sx, int pz, int py, int px,
                         int kd, int kh, int kw) {
  int i = blockIdx.x * blockDim.x + threadIdx.x;
  int no = Do * Ho * Wo;
  if (i >= no) return;
  int x = i % Wo, y = (i / Wo) % Ho, z = i / (Wo * Ho);
  unsigned char act = 0;
  for (int a = 0; a < kd; ++a) {
    int iz = z * sz - pz + a; if ((unsigned)iz >= (unsigned)Di) continue;
    for (int b = 0; b < kh; ++b) {
      int iy = y * sy - py + b; if ((unsigned)iy >= (unsigned)Hi) continue;
      for (int c = 0; c < kw; ++c) {
        int ix = x * sx - px + c; if ((unsigned)ix >= (unsigned)Wi) continue;
        act |= mi[((long long)iz * Hi + iy) * Wi + ix];
      }
    }
  }
  mo[i] = act ? 1 : 0;
}

// fold BN into scale/bias: y = relu(h*scale + bias)
__global__ void k_bnprep(const float* g, const float* b, const float* m, const float* v,
                         float* scale, float* bias, int cout) {
  int i = blockIdx.x * blockDim.x + threadIdx.x;
  if (i >= cout) return;
  float s = g[i] * rsqrtf(v[i] + 1e-3f);
  scale[i] = s;
  bias[i]  = b[i] - m[i] * s;
}

// pack f32 weights (tap, cin, cout) into wave32 B-fragment bf16 layout:
//   [tap][kTile][coutTile][lane 0..31][16 bf16]; lane&15 = N column,
//   lane>>4 selects K half {0..7,16..23} vs {8..15,24..31}.
__global__ void k_packw(const float* __restrict__ w, int taps, int cin, int cout,
                        unsigned short* __restrict__ wp) {
  int nK = (cin + 31) >> 5, nN = cout >> 4;
  int total = taps * nK * nN * 32;
  int t = blockIdx.x * blockDim.x + threadIdx.x;
  if (t >= total) return;
  int lane = t & 31; int rest = t >> 5;
  int nIdx = rest % nN; rest /= nN;
  int kIdx = rest % nK; int tap = rest / nK;
  int halfK = lane >> 4, n = lane & 15;
  unsigned short* dst = wp + (size_t)t * 16;
  for (int j = 0; j < 16; ++j) {
    int kk = kIdx * 32 + ((j < 8) ? (halfK * 8 + j) : (16 + halfK * 8 + (j - 8)));
    float val = (kk < cin) ? w[((size_t)tap * cin + kk) * cout + (nIdx * 16 + n)] : 0.f;
    dst[j] = f2bf(val);
  }
}

// -------------------- WMMA gather-GEMM sparse conv layer --------------------
// One wave handles one tile of 16 active output voxels x ALL cout tiles.
// NK = K tiles of 32 (cin), NN = cout/16; both compile-time so the NN f32
// accumulator fragments and the kt/n loops fully unroll. EXEC stays full
// across WMMA (guards are on loads only; OOB/padded-K/tail rows give zero A).
template <int NK, int NN, bool FINAL>
__global__ __launch_bounds__(128)
void k_sconv(const unsigned short* __restrict__ tin, int Di, int Hi, int Wi,
             int cinPad,
             unsigned short* __restrict__ tout, float* __restrict__ fout,
             int Wo, int Ho,
             const unsigned short* __restrict__ wp,
             const float* __restrict__ scale, const float* __restrict__ bias,
             const int* __restrict__ list, const int* __restrict__ count,
             int sz, int sy, int sx, int pz, int py, int px,
             int kd, int kh, int kw) {
  const int lane = threadIdx.x & 31;
  const int wave = threadIdx.x >> 5;
  const int cnt = *count;
  const int tile = blockIdx.x * 4 + wave;
  const int nTiles = (cnt + 15) >> 4;
  if (tile >= nTiles) return;

  const int m = lane & 15;          // A row (output voxel within tile)
  const int halfK = lane >> 4;      // which K half this lane carries
  const int vi = tile * 16 + m;
  const bool rowv = vi < cnt;
  const int ov = rowv ? list[vi] : 0;
  const int ox = ov % Wo;
  const int oy = (ov / Wo) % Ho;
  const int oz = ov / (Wo * Ho);
  const int taps = kd * kh * kw;
  const uint4 z4 = make_uint4(0u, 0u, 0u, 0u);

  v8f acc[NN];
#pragma unroll
  for (int n = 0; n < NN; ++n) acc[n] = (v8f){0.f, 0.f, 0.f, 0.f, 0.f, 0.f, 0.f, 0.f};

  for (int tap = 0; tap < taps; ++tap) {
    int a = tap / (kh * kw);
    int rem = tap - a * (kh * kw);
    int b = rem / kw;
    int c = rem - b * kw;
    int iz = oz * sz - pz + a;
    int iy = oy * sy - py + b;
    int ix = ox * sx - px + c;
    bool in = rowv && (unsigned)iz < (unsigned)Di && (unsigned)iy < (unsigned)Hi &&
              (unsigned)ix < (unsigned)Wi;
    const unsigned short* src =
        tin + (in ? (((size_t)iz * Hi + iy) * Wi + ix) * (size_t)cinPad : (size_t)0);
    const unsigned short* wtap = wp + (size_t)tap * NK * NN * 512 + lane * 16;
#pragma unroll
    for (int kt = 0; kt < NK; ++kt) {
      union { uint4 q[2]; v16bf v; } A;
      int c0 = kt * 32 + halfK * 8;      // K   0..7 / 8..15
      int c1 = c0 + 16;                  // K 16..23 / 24..31
      A.q[0] = (in && (c0 + 8) <= cinPad) ? *reinterpret_cast<const uint4*>(src + c0) : z4;
      A.q[1] = (in && (c1 + 8) <= cinPad) ? *reinterpret_cast<const uint4*>(src + c1) : z4;
#pragma unroll
      for (int n = 0; n < NN; ++n) {
        union { uint4 q[2]; v16bf v; } B;
        const unsigned short* bp = wtap + (size_t)(kt * NN + n) * 512;
        B.q[0] = *reinterpret_cast<const uint4*>(bp);
        B.q[1] = *reinterpret_cast<const uint4*>(bp + 8);
        acc[n] = __builtin_amdgcn_wmma_f32_16x16x32_bf16(false, A.v, false, B.v,
                                                         (short)0, acc[n], false, false);
      }
    }
  }

  // Epilogue: C/D layout — lanes 0..15 hold rows r, lanes 16..31 rows r+8.
  const int nc = lane & 15;
#pragma unroll
  for (int n = 0; n < NN; ++n) {
    const int co = n * 16 + nc;
    const float s = scale[co], bi = bias[co];
#pragma unroll
    for (int r = 0; r < 8; ++r) {
      int row = (lane < 16) ? r : (r + 8);
      int vr = tile * 16 + row;
      if (vr < cnt) {
        int ovr = list[vr];
        float o = acc[n][r] * s + bi;
        o = o > 0.f ? o : 0.f;
        if constexpr (FINAL) fout[(size_t)ovr * (NN * 16) + co] = o;
        else                 tout[(size_t)ovr * (NN * 16) + co] = f2bf(o);
      }
    }
  }
}

// --------------------------------- host ------------------------------------

extern "C" void kernel_launch(void* const* d_in, const int* in_sizes, int n_in,
                              void* d_out, int out_size, void* d_ws, size_t ws_size,
                              hipStream_t stream) {
  (void)in_sizes; (void)n_in; (void)out_size; (void)ws_size;
  // Stage grids
  const int D0 = 41, H0 = 400, W0 = 352;   // stage 0
  const int D1 = 21, H1 = 200, W1 = 176;   // after L2
  const int D2 = 11, H2 = 100, W2 = 88;    // after L5
  const int D3 = 5,  H3 = 50,  W3 = 44;    // after L8
  const int D4 = 2,  H4 = 50,  W4 = 44;    // after L11
  const long long NV0 = (long long)D0 * H0 * W0;  // 5,772,800
  const long long NV1 = (long long)D1 * H1 * W1;  //   739,200
  const long long NV2 = (long long)D2 * H2 * W2;  //    96,800
  const long long NV3 = (long long)D3 * H3 * W3;  //    11,000
  const long long NV4 = (long long)D4 * H4 * W4;  //     4,400
  const int NIN = 100000;

  const float* feats  = (const float*)d_in[0];
  const int*   coords = (const int*)d_in[1];
  const float *Wm[12], *Gm[12], *Bm[12], *Mm[12], *Vm[12];
  for (int i = 0; i < 12; ++i) {
    Wm[i] = (const float*)d_in[2 + 5 * i + 0];
    Gm[i] = (const float*)d_in[2 + 5 * i + 1];
    Bm[i] = (const float*)d_in[2 + 5 * i + 2];
    Mm[i] = (const float*)d_in[2 + 5 * i + 3];
    Vm[i] = (const float*)d_in[2 + 5 * i + 4];
  }

  // ---- workspace layout ----
  char* ws = (char*)d_ws;
  size_t off = 0;
  auto take = [&](size_t bytes) { size_t o = off; off += (bytes + 255) & ~(size_t)255; return o; };
  size_t o_regA = take((size_t)NV0 * 8 * 2);      // t_in (bf16, cpad 8); reused below
  size_t o_tA   = take((size_t)NV0 * 16 * 2);     // stage0 ping (16 ch bf16)
  size_t o_tB   = take((size_t)NV0 * 16 * 2);     // stage0 pong
  size_t o_slot = take((size_t)NV0 * 4);
  size_t o_m0 = take(NV0), o_m1 = take(NV1), o_m2 = take(NV2), o_m3 = take(NV3), o_m4 = take(NV4);
  size_t o_l0 = take((size_t)NIN * 4), o_l1 = take((size_t)NV1 * 4),
         o_l2 = take((size_t)NV2 * 4), o_l3 = take((size_t)NV3 * 4), o_l4 = take((size_t)NV4 * 4);
  size_t o_cnt = take(64);
  size_t o_bn  = take(12 * 1024);                 // per layer: 128 scale f32 + 128 bias f32
  static const size_t wpsz[12] = {27648, 27648, 55296, 55296, 55296, 110592,
                                  221184, 221184, 221184, 221184, 221184, 49152};
  size_t o_wp[12];
  for (int i = 0; i < 12; ++i) o_wp[i] = take(wpsz[i]);
  // Overlap later-stage buffers into regionA (t_in dead after L0):
  size_t o_s1a = o_regA + 0;                        // 47,308,800 B
  size_t o_s2a = o_regA + (size_t)48 * 1024 * 1024; // 12,390,400 B
  size_t o_s2b = o_regA + (size_t)61 * 1024 * 1024;
  size_t o_s3a = o_regA + (size_t)75 * 1024 * 1024; //  1,408,000 B
  size_t o_s3b = o_regA + (size_t)78 * 1024 * 1024;
  size_t o_s1b = o_tA;                              // t_a dead after L1

  auto zero = [&](size_t ob, size_t bytes) {
    long long n16 = (long long)((bytes + 15) / 16);
    unsigned blocks = (unsigned)((n16 + 255) / 256);
    k_zero<<<dim3(blocks), 256, 0, stream>>>((uint4*)(ws + ob), n16);
  };
  // NK/NN select the template instantiation (compile-time unrolled loops).
  auto conv = [&](size_t o_in, int Di, int Hi, int Wi, int cinPad,
                  size_t o_out, float* fout, int Ho_, int Wo_,
                  int li, size_t o_list, int cslot, int maxCnt,
                  int sz, int sy, int sx, int pz, int py, int px,
                  int kd, int kh, int kw, int NK, int NN) {
    int maxTiles = (maxCnt + 15) / 16;
    dim3 g((unsigned)((maxTiles + 3) / 4), 1, 1);
    const unsigned short* tin = (const unsigned short*)(ws + o_in);
    unsigned short* tout = fout ? nullptr : (unsigned short*)(ws + o_out);
    const unsigned short* wpck = (const unsigned short*)(ws + o_wp[li]);
    const float* sc = (const float*)(ws + o_bn + (size_t)li * 1024);
    const float* bi = (const float*)(ws + o_bn + (size_t)li * 1024 + 512);
    const int* lst = (const int*)(ws + o_list);
    const int* cp  = (const int*)(ws + o_cnt) + cslot;
    if (fout) {
      k_sconv<2, 8, true><<<g, 128, 0, stream>>>(tin, Di, Hi, Wi, cinPad, tout, fout,
          Wo_, Ho_, wpck, sc, bi, lst, cp, sz, sy, sx, pz, py, px, kd, kh, kw);
    } else if (NK == 1 && NN == 1) {
      k_sconv<1, 1, false><<<g, 128, 0, stream>>>(tin, Di, Hi, Wi, cinPad, tout, fout,
          Wo_, Ho_, wpck, sc, bi, lst, cp, sz, sy, sx, pz, py, px, kd, kh, kw);
    } else if (NK == 1 && NN == 2) {
      k_sconv<1, 2, false><<<g, 128, 0, stream>>>(tin, Di, Hi, Wi, cinPad, tout, fout,
          Wo_, Ho_, wpck, sc, bi, lst, cp, sz, sy, sx, pz, py, px, kd, kh, kw);
    } else if (NK == 1 && NN == 4) {
      k_sconv<1, 4, false><<<g, 128, 0, stream>>>(tin, Di, Hi, Wi, cinPad, tout, fout,
          Wo_, Ho_, wpck, sc, bi, lst, cp, sz, sy, sx, pz, py, px, kd, kh, kw);
    } else {
      k_sconv<2, 4, false><<<g, 128, 0, stream>>>(tin, Di, Hi, Wi, cinPad, tout, fout,
          Wo_, Ho_, wpck, sc, bi, lst, cp, sz, sy, sx, pz, py, px, kd, kh, kw);
    }
  };

  // ---- init: zero scatter targets, counters ----
  zero(o_slot, (size_t)NV0 * 4);
  zero(o_m0, NV0);
  zero(o_regA, (size_t)NV0 * 8 * 2);   // t_in
  zero(o_cnt, 64);

  // ---- BN fold + weight packing (L2-resident, ~1.4 MB total) ----
  static const int Lcin[12]  = {4, 16, 16, 32, 32, 32, 64, 64, 64, 64, 64, 64};
  static const int Lcout[12] = {16, 16, 32, 32, 32, 64, 64, 64, 64, 64, 64, 128};
  static const int Ltaps[12] = {27, 27, 27, 27, 27, 27, 27, 27, 27, 27, 27, 3};
  for (int i = 0; i < 12; ++i) {
    k_bnprep<<<1, 128, 0, stream>>>(Gm[i], Bm[i], Mm[i], Vm[i],
                                    (float*)(ws + o_bn + (size_t)i * 1024),
                                    (float*)(ws + o_bn + (size_t)i * 1024 + 512), Lcout[i]);
    int total = Ltaps[i] * ((Lcin[i] + 31) / 32) * (Lcout[i] / 16) * 32;
    k_packw<<<(total + 127) / 128, 128, 0, stream>>>(Wm[i], Ltaps[i], Lcin[i], Lcout[i],
                                                     (unsigned short*)(ws + o_wp[i]));
  }

  // ---- scatter inputs (dedup), build stage-0 active list ----
  k_scatter1<<<(NIN + 255) / 256, 256, 0, stream>>>(coords, (int*)(ws + o_slot), NIN, H0, W0);
  k_scatter2<<<(NIN + 255) / 256, 256, 0, stream>>>(feats, coords, (const int*)(ws + o_slot),
                                                    (unsigned short*)(ws + o_regA),
                                                    (unsigned char*)(ws + o_m0), NIN, H0, W0, 8);
  k_build_list<<<(unsigned)((NV0 + 255) / 256), 256, 0, stream>>>(
      (const unsigned char*)(ws + o_m0), (int)NV0, (int*)(ws + o_l0), (int*)(ws + o_cnt) + 0);

  // ---- stage 0: L0, L1 (subm, ~100k active of 5.77M) ----
  zero(o_tA, (size_t)NV0 * 16 * 2);
  conv(o_regA, D0, H0, W0, 8,  o_tA, nullptr, H0, W0, 0, o_l0, 0, NIN, 1,1,1, 1,1,1, 3,3,3, 1, 1);
  zero(o_tB, (size_t)NV0 * 16 * 2);
  conv(o_tA,  D0, H0, W0, 16, o_tB, nullptr, H0, W0, 1, o_l0, 0, NIN, 1,1,1, 1,1,1, 3,3,3, 1, 1);

  // ---- stage 1: L2 spconv s2, then L3/L4 subm ----
  k_dilate<<<(unsigned)((NV1 + 255) / 256), 256, 0, stream>>>(
      (const unsigned char*)(ws + o_m0), D0, H0, W0,
      (unsigned char*)(ws + o_m1), D1, H1, W1, 2,2,2, 1,1,1, 3,3,3);
  k_build_list<<<(unsigned)((NV1 + 255) / 256), 256, 0, stream>>>(
      (const unsigned char*)(ws + o_m1), (int)NV1, (int*)(ws + o_l1), (int*)(ws + o_cnt) + 1);
  zero(o_s1a, (size_t)NV1 * 32 * 2);
  conv(o_tB,  D0, H0, W0, 16, o_s1a, nullptr, H1, W1, 2, o_l1, 1, (int)NV1, 2,2,2, 1,1,1, 3,3,3, 1, 2);
  zero(o_s1b, (size_t)NV1 * 32 * 2);
  conv(o_s1a, D1, H1, W1, 32, o_s1b, nullptr, H1, W1, 3, o_l1, 1, (int)NV1, 1,1,1, 1,1,1, 3,3,3, 1, 2);
  zero(o_s1a, (size_t)NV1 * 32 * 2);
  conv(o_s1b, D1, H1, W1, 32, o_s1a, nullptr, H1, W1, 4, o_l1, 1, (int)NV1, 1,1,1, 1,1,1, 3,3,3, 1, 2);

  // ---- stage 2: L5 spconv s2, then L6/L7 subm ----
  k_dilate<<<(unsigned)((NV2 + 255) / 256), 256, 0, stream>>>(
      (const unsigned char*)(ws + o_m1), D1, H1, W1,
      (unsigned char*)(ws + o_m2), D2, H2, W2, 2,2,2, 1,1,1, 3,3,3);
  k_build_list<<<(unsigned)((NV2 + 255) / 256), 256, 0, stream>>>(
      (const unsigned char*)(ws + o_m2), (int)NV2, (int*)(ws + o_l2), (int*)(ws + o_cnt) + 2);
  zero(o_s2a, (size_t)NV2 * 64 * 2);
  conv(o_s1a, D1, H1, W1, 32, o_s2a, nullptr, H2, W2, 5, o_l2, 2, (int)NV2, 2,2,2, 1,1,1, 3,3,3, 1, 4);
  zero(o_s2b, (size_t)NV2 * 64 * 2);
  conv(o_s2a, D2, H2, W2, 64, o_s2b, nullptr, H2, W2, 6, o_l2, 2, (int)NV2, 1,1,1, 1,1,1, 3,3,3, 2, 4);
  zero(o_s2a, (size_t)NV2 * 64 * 2);
  conv(o_s2b, D2, H2, W2, 64, o_s2a, nullptr, H2, W2, 7, o_l2, 2, (int)NV2, 1,1,1, 1,1,1, 3,3,3, 2, 4);

  // ---- stage 3: L8 spconv s2 pad(0,1,1), then L9/L10 subm ----
  k_dilate<<<(unsigned)((NV3 + 255) / 256), 256, 0, stream>>>(
      (const unsigned char*)(ws + o_m2), D2, H2, W2,
      (unsigned char*)(ws + o_m3), D3, H3, W3, 2,2,2, 0,1,1, 3,3,3);
  k_build_list<<<(unsigned)((NV3 + 255) / 256), 256, 0, stream>>>(
      (const unsigned char*)(ws + o_m3), (int)NV3, (int*)(ws + o_l3), (int*)(ws + o_cnt) + 3);
  zero(o_s3a, (size_t)NV3 * 64 * 2);
  conv(o_s2a, D2, H2, W2, 64, o_s3a, nullptr, H3, W3, 8, o_l3, 3, (int)NV3, 2,2,2, 0,1,1, 3,3,3, 2, 4);
  zero(o_s3b, (size_t)NV3 * 64 * 2);
  conv(o_s3a, D3, H3, W3, 64, o_s3b, nullptr, H3, W3, 9, o_l3, 3, (int)NV3, 1,1,1, 1,1,1, 3,3,3, 2, 4);
  zero(o_s3a, (size_t)NV3 * 64 * 2);
  conv(o_s3b, D3, H3, W3, 64, o_s3a, nullptr, H3, W3, 10, o_l3, 3, (int)NV3, 1,1,1, 1,1,1, 3,3,3, 2, 4);

  // ---- stage 4: L11 spconv k(3,1,1) s(2,1,1) p0 -> f32 d_out ----
  k_dilate<<<(unsigned)((NV4 + 255) / 256), 256, 0, stream>>>(
      (const unsigned char*)(ws + o_m3), D3, H3, W3,
      (unsigned char*)(ws + o_m4), D4, H4, W4, 2,1,1, 0,0,0, 3,1,1);
  k_build_list<<<(unsigned)((NV4 + 255) / 256), 256, 0, stream>>>(
      (const unsigned char*)(ws + o_m4), (int)NV4, (int*)(ws + o_l4), (int*)(ws + o_cnt) + 4);
  { // zero d_out (2,252,800 bytes, /16 exact)
    long long n16 = ((long long)NV4 * 128 * 4) / 16;
    k_zero<<<(unsigned)((n16 + 255) / 256), 256, 0, stream>>>((uint4*)d_out, n16);
  }
  conv(o_s3a, D3, H3, W3, 64, 0, (float*)d_out, H4, W4, 11, o_l4, 4, (int)NV4,
       2,1,1, 0,0,0, 3,1,1, 2, 8);
}
